// VGAE_Classifier_86320252715258
// MI455X (gfx1250) — matrix operand order repaired and checked by reading, not compiled, not run on hardware
//
#include <hip/hip_runtime.h>
#include <hip/hip_bf16.h>

#define N_NODES 50000
#define N_EDGES 1600000
#define BGR 64
#define INF_DIM 16
#define HDIM 128
#define LDIM 64
#define CDIM 6

typedef __attribute__((ext_vector_type(16))) _Float16 v16h;
typedef __attribute__((ext_vector_type(8)))  float    v8f;

// ---------------- utility ----------------
__global__ void k_zero(float* p, int n) {
  int i = blockIdx.x * blockDim.x + threadIdx.x;
  if (i < n) p[i] = 0.f;
}

// ---------------- degree / norm ----------------
__global__ void k_deg(const int* dst, const float* ew, float* deg) {
  int e = blockIdx.x * blockDim.x + threadIdx.x;
  if (e < N_EDGES) atomicAdd(&deg[dst[e]], ew[e]);
}

__global__ void k_dinv(float* deg) {
  int i = blockIdx.x * blockDim.x + threadIdx.x;
  if (i < N_NODES) deg[i] = rsqrtf(deg[i] + 1.0f);
}

__global__ void k_norm(const int* src, const int* dst, const float* ew,
                       const float* dinv, float* norm) {
  int e = blockIdx.x * blockDim.x + threadIdx.x;
  if (e < N_EDGES) norm[e] = ew[e] * dinv[src[e]] * dinv[dst[e]];
}

// ---------------- weight pack: f32 [128][ncols] -> WMMA-lane-layout f16 ----------------
// packed index: (((ks*ntiles + nt)*32 + lane)*16 + e)
// value = W[ks*32 + (e>>3)*16 + (lane>>4)*8 + (e&7)][nt*16 + (lane&15)]
__global__ void k_pack_w(const float* W, _Float16* Bp, int ncols) {
  int idx = blockIdx.x * blockDim.x + threadIdx.x;
  int total = HDIM * ncols;
  if (idx >= total) return;
  int e = idx & 15;
  int lane = (idx >> 4) & 31;
  int t = idx >> 9;                 // ks*ntiles + nt
  int ntiles = ncols >> 4;
  int ks = t / ntiles, nt = t - ks * ntiles;
  int k = ks * 32 + ((e >> 3) << 4) + ((lane >> 4) << 3) + (e & 7);
  int n = (nt << 4) + (lane & 15);
  Bp[idx] = (_Float16)W[k * ncols + n];
}

// ---------------- input linear: h = x @ W_in + b_in  ([N,16]x[16,128]) ----------------
__global__ void k_lin_in(const float* __restrict__ x, const float* __restrict__ W,
                         const float* __restrict__ b, float* __restrict__ h) {
  int idx = blockIdx.x * blockDim.x + threadIdx.x;
  if (idx >= N_NODES * HDIM) return;
  int node = idx >> 7, c = idx & 127;
  const float* xr = x + node * INF_DIM;
  float s = b[c];
#pragma unroll
  for (int k = 0; k < INF_DIM; ++k) s += xr[k] * W[k * HDIM + c];
  h[idx] = s;
}

// ---------------- WMMA GEMM: C[N][ncols] = A[N][128] @ Wpacked + bias ----------------
// One wave computes a 16x64 output strip: 4 accumulators share each A fragment, so
// A is loaded/converted once per K-step and 4 v_wmma issue back-to-back (no D->A/B
// hazard: independent accumulators).
__global__ void k_gemm_wmma(const float* __restrict__ A, const _Float16* __restrict__ Bp,
                            const float* __restrict__ bias, float* __restrict__ C, int ncols) {
  int lane = threadIdx.x & 31;
  int wave = threadIdx.x >> 5;
  int ntiles = ncols >> 4;          // 8 (HDIM) or 4 (LDIM)
  int ngroups = ntiles >> 2;        // groups of 4 n-tiles
  int job = blockIdx.x * (blockDim.x >> 5) + wave;
  int totalJobs = (N_NODES >> 4) * ngroups;
  if (job >= totalJobs) return;     // wave-uniform: EXEC stays all-ones
  int mt = job / ngroups;
  int nt0 = (job - mt * ngroups) << 2;
  int row0 = mt << 4;
  int m = lane & 15, kh = lane >> 4;
  const float* arow = A + (size_t)(row0 + m) * HDIM;

  v8f acc0 = {}, acc1 = {}, acc2 = {}, acc3 = {};
#pragma unroll
  for (int ks = 0; ks < 4; ++ks) {
    if (ks < 3)
      __builtin_prefetch(arow + (ks + 1) * 32 + kh * 8, 0, 3);  // global_prefetch_b8
    // A 16x32 f16 lane layout: element e -> K = ks*32 + g*16 + kh*8 + w (e = g*8+w)
    v16h a;
#pragma unroll
    for (int g = 0; g < 2; ++g) {
      const float* p = arow + ks * 32 + g * 16 + kh * 8;
#pragma unroll
      for (int w = 0; w < 8; ++w) a[g * 8 + w] = (_Float16)p[w];
    }
    // B 32x16 f16: packed so each lane loads 16 contiguous halves (32B) per tile
    const _Float16* bb = Bp + (size_t)((ks * ntiles + nt0) * 32 + lane) * 16;
    v16h b0 = *(const v16h*)(bb);
    v16h b1 = *(const v16h*)(bb + 32 * 16);
    v16h b2 = *(const v16h*)(bb + 64 * 16);
    v16h b3 = *(const v16h*)(bb + 96 * 16);
    acc0 = __builtin_amdgcn_wmma_f32_16x16x32_f16(false, a, false, b0, (short)0, acc0, false, false);
    acc1 = __builtin_amdgcn_wmma_f32_16x16x32_f16(false, a, false, b1, (short)0, acc1, false, false);
    acc2 = __builtin_amdgcn_wmma_f32_16x16x32_f16(false, a, false, b2, (short)0, acc2, false, false);
    acc3 = __builtin_amdgcn_wmma_f32_16x16x32_f16(false, a, false, b3, (short)0, acc3, false, false);
  }
  // C/D layout: VGPR r -> row = row0 + kh*8 + r, col = (nt0+q)*16 + (lane&15)
#pragma unroll
  for (int q = 0; q < 4; ++q) {
    v8f acc = (q == 0) ? acc0 : (q == 1) ? acc1 : (q == 2) ? acc2 : acc3;
    int col = ((nt0 + q) << 4) + m;
    float bv = bias[col];
#pragma unroll
    for (int r = 0; r < 8; ++r)
      C[(size_t)(row0 + kh * 8 + r) * ncols + col] = acc[r] + bv;
  }
}

// ---------------- edge scatter: agg[dst] += h[src] * norm ----------------
__global__ void k_scatter(const float* __restrict__ H, const int* __restrict__ src,
                          const int* __restrict__ dst, const float* __restrict__ norm,
                          float* __restrict__ agg, int ncols) {
  long long idx = (long long)blockIdx.x * blockDim.x + threadIdx.x;
  int perEdge = ncols >> 2;
  long long total = (long long)N_EDGES * perEdge;
  if (idx >= total) return;
  int e = (int)(idx / perEdge);
  int c4 = (int)(idx - (long long)e * perEdge) << 2;
  int s = src[e], d = dst[e];
  float nv = norm[e];
  const float4 hv = *(const float4*)(H + (size_t)s * ncols + c4);
  float* o = agg + (size_t)d * ncols + c4;
  atomicAdd(o + 0, hv.x * nv);
  atomicAdd(o + 1, hv.y * nv);
  atomicAdd(o + 2, hv.z * nv);
  atomicAdd(o + 3, hv.w * nv);
}

// ---------------- GCN epilogue: self-loop + BN(eval) + ReLU ----------------
__global__ void k_bn_relu(const float* __restrict__ gm, const float* __restrict__ agg,
                          const float* __restrict__ dinv, const float* __restrict__ gg,
                          const float* __restrict__ be, const float* __restrict__ rm,
                          const float* __restrict__ rv, float* __restrict__ out) {
  int idx = blockIdx.x * blockDim.x + threadIdx.x;
  if (idx >= N_NODES * HDIM) return;
  int node = idx >> 7, c = idx & 127;
  float di = dinv[node];
  float v = agg[idx] + gm[idx] * di * di;
  float sc = gg[c] * rsqrtf(rv[c] + 1e-5f);
  out[idx] = fmaxf((v - rm[c]) * sc + be[c], 0.f);
}

// self-loop only (mu / logvar heads), ncols = LDIM
__global__ void k_selfloop(const float* __restrict__ gm, const float* __restrict__ agg,
                           const float* __restrict__ dinv, float* __restrict__ out) {
  int idx = blockIdx.x * blockDim.x + threadIdx.x;
  if (idx >= N_NODES * LDIM) return;
  int node = idx >> 6;
  float di = dinv[node];
  out[idx] = agg[idx] + gm[idx] * di * di;
}

// ---------------- attention scores: tanh(z@Wa1+ba1)@Wa2 + ba2 ----------------
__global__ void k_att(const float* __restrict__ mu, const float* __restrict__ Wa1,
                      const float* __restrict__ ba1, const float* __restrict__ Wa2,
                      const float* __restrict__ ba2, float* __restrict__ score) {
  __shared__ float zsh[LDIM];
  __shared__ float red[HDIM];
  int n = blockIdx.x, t = threadIdx.x;
  if (t < LDIM) zsh[t] = mu[n * LDIM + t];
  __syncthreads();
  float s = ba1[t];
  for (int k = 0; k < LDIM; ++k) s += zsh[k] * Wa1[k * HDIM + t];
  red[t] = tanhf(s) * Wa2[t];
  __syncthreads();
  for (int off = 64; off > 0; off >>= 1) {
    if (t < off) red[t] += red[t + off];
    __syncthreads();
  }
  if (t == 0) score[n] = red[0] + ba2[0];
}

// ---------------- global softmax reductions ----------------
__global__ void k_rmax(const float* __restrict__ v, int n, float* __restrict__ out) {
  __shared__ float sm[256];
  float m = -3.402823466e38f;
  for (int i = blockIdx.x * blockDim.x + threadIdx.x; i < n; i += gridDim.x * blockDim.x)
    m = fmaxf(m, v[i]);
  sm[threadIdx.x] = m;
  __syncthreads();
  for (int off = 128; off > 0; off >>= 1) {
    if (threadIdx.x < off) sm[threadIdx.x] = fmaxf(sm[threadIdx.x], sm[threadIdx.x + off]);
    __syncthreads();
  }
  if (threadIdx.x == 0) out[blockIdx.x] = sm[0];
}

__global__ void k_rsumexp(const float* __restrict__ v, int n, const float* __restrict__ mx,
                          float* __restrict__ out) {
  __shared__ float sm[256];
  float m = mx[0], s = 0.f;
  for (int i = blockIdx.x * blockDim.x + threadIdx.x; i < n; i += gridDim.x * blockDim.x)
    s += __expf(v[i] - m);
  sm[threadIdx.x] = s;
  __syncthreads();
  for (int off = 128; off > 0; off >>= 1) {
    if (threadIdx.x < off) sm[threadIdx.x] += sm[threadIdx.x + off];
    __syncthreads();
  }
  if (threadIdx.x == 0) out[blockIdx.x] = sm[0];
}

__global__ void k_rsum(const float* __restrict__ v, int n, float* __restrict__ out) {
  __shared__ float sm[256];
  float s = 0.f;
  for (int i = blockIdx.x * blockDim.x + threadIdx.x; i < n; i += gridDim.x * blockDim.x)
    s += v[i];
  sm[threadIdx.x] = s;
  __syncthreads();
  for (int off = 128; off > 0; off >>= 1) {
    if (threadIdx.x < off) sm[threadIdx.x] += sm[threadIdx.x + off];
    __syncthreads();
  }
  if (threadIdx.x == 0) out[blockIdx.x] = sm[0];
}

// ---------------- pooling: attention-weighted sum + mean-pool accumulators ----------------
__global__ void k_pool(const float* __restrict__ mu, const float* __restrict__ score,
                       const int* __restrict__ batch, const float* __restrict__ stats,
                       float* __restrict__ g, float* __restrict__ zsum, float* __restrict__ cnt) {
  int idx = blockIdx.x * blockDim.x + threadIdx.x;
  if (idx >= N_NODES * LDIM) return;
  int n = idx >> 6, l = idx & 63;
  float att = __expf(score[n] - stats[0]) / stats[1];
  float zv = mu[idx];
  int b = batch[n];
  atomicAdd(&g[b * LDIM + l], zv * att);
  atomicAdd(&zsum[b * LDIM + l], zv);
  if (l == 0) atomicAdd(&cnt[b], 1.0f);
}

// ---------------- per-graph heads: classifier MLP + noise detector ----------------
__global__ void k_head(const float* __restrict__ g, const float* __restrict__ zsum,
                       const float* __restrict__ cnt,
                       const float* Wc1, const float* bc1, const float* Wc2, const float* bc2,
                       const float* Wc3, const float* bc3,
                       const float* Wn1, const float* bn1, const float* Wn2, const float* bn2,
                       float* __restrict__ pred, float* __restrict__ noise) {
  __shared__ float gb[LDIM], p1[HDIM], p2[LDIM], zm[LDIM], red[HDIM];
  int b = blockIdx.x, t = threadIdx.x;
  if (t < LDIM) {
    gb[t] = g[b * LDIM + t];
    zm[t] = zsum[b * LDIM + t] / fmaxf(cnt[b], 1.f);
  }
  __syncthreads();
  {
    float s = bc1[t];
    for (int k = 0; k < LDIM; ++k) s += gb[k] * Wc1[k * HDIM + t];
    p1[t] = fmaxf(s, 0.f);
  }
  __syncthreads();
  if (t < LDIM) {
    float s = bc2[t];
    for (int k = 0; k < HDIM; ++k) s += p1[k] * Wc2[k * LDIM + t];
    p2[t] = fmaxf(s, 0.f);
  }
  __syncthreads();
  if (t < CDIM) {
    float s = bc3[t];
    for (int k = 0; k < LDIM; ++k) s += p2[k] * Wc3[k * CDIM + t];
    pred[b * CDIM + t] = s;
  }
  if (t < LDIM) {
    float s = bn1[t];
    for (int k = 0; k < LDIM; ++k) s += zm[k] * Wn1[k * LDIM + t];
    red[t] = fmaxf(s, 0.f) * Wn2[t];
  } else {
    red[t] = 0.f;
  }
  __syncthreads();
  for (int off = 64; off > 0; off >>= 1) {
    if (t < off) red[t] += red[t + off];
    __syncthreads();
  }
  if (t == 0) noise[b] = 1.f / (1.f + __expf(-(red[0] + bn2[0])));
}

// ---------------- host ----------------
extern "C" void kernel_launch(void* const* d_in, const int* in_sizes, int n_in,
                              void* d_out, int out_size, void* d_ws, size_t ws_size,
                              hipStream_t stream) {
  (void)in_sizes; (void)n_in; (void)out_size; (void)ws_size;
  const float* x    = (const float*)d_in[0];
  const int*   ei   = (const int*)d_in[1];
  const float* ea   = (const float*)d_in[2];
  const int*   batch= (const int*)d_in[3];
  const float* W_in = (const float*)d_in[4];  const float* b_in = (const float*)d_in[5];
  const float* W1   = (const float*)d_in[6];  const float* b1   = (const float*)d_in[7];
  const float* W2   = (const float*)d_in[8];  const float* b2   = (const float*)d_in[9];
  const float* Wmu  = (const float*)d_in[10]; const float* bmu  = (const float*)d_in[11];
  const float* Wlv  = (const float*)d_in[12]; const float* blv  = (const float*)d_in[13];
  const float* g1   = (const float*)d_in[14]; const float* be1  = (const float*)d_in[15];
  const float* rm1  = (const float*)d_in[16]; const float* rv1  = (const float*)d_in[17];
  const float* g2   = (const float*)d_in[18]; const float* be2  = (const float*)d_in[19];
  const float* rm2  = (const float*)d_in[20]; const float* rv2  = (const float*)d_in[21];
  const float* Wa1  = (const float*)d_in[22]; const float* ba1  = (const float*)d_in[23];
  const float* Wa2  = (const float*)d_in[24]; const float* ba2  = (const float*)d_in[25];
  const float* Wc1  = (const float*)d_in[26]; const float* bc1  = (const float*)d_in[27];
  const float* Wc2  = (const float*)d_in[28]; const float* bc2  = (const float*)d_in[29];
  const float* Wc3  = (const float*)d_in[30]; const float* bc3  = (const float*)d_in[31];
  const float* Wn1  = (const float*)d_in[32]; const float* bn1  = (const float*)d_in[33];
  const float* Wn2  = (const float*)d_in[34]; const float* bn2  = (const float*)d_in[35];

  const int* src = ei;
  const int* dst = ei + N_EDGES;

  char* ws = (char*)d_ws;
  size_t off = 0;
  auto alloc = [&](size_t bytes) -> char* {
    char* p = ws + off;
    off = (off + bytes + 255) & ~(size_t)255;
    return p;
  };
  float*     deg  = (float*)alloc((size_t)N_NODES * 4);          // becomes dinv in place
  float*     norm = (float*)alloc((size_t)N_EDGES * 4);
  float*     hA   = (float*)alloc((size_t)N_NODES * HDIM * 4);   // activations
  float*     hB   = (float*)alloc((size_t)N_NODES * HDIM * 4);   // gemm output
  float*     hC   = (float*)alloc((size_t)N_NODES * HDIM * 4);   // aggregation
  _Float16*  W1p  = (_Float16*)alloc((size_t)HDIM * HDIM * 2);
  _Float16*  W2p  = (_Float16*)alloc((size_t)HDIM * HDIM * 2);
  _Float16*  Wmup = (_Float16*)alloc((size_t)HDIM * LDIM * 2);
  _Float16*  Wlvp = (_Float16*)alloc((size_t)HDIM * LDIM * 2);
  float*     score= (float*)alloc((size_t)N_NODES * 4);
  float*     pmax = (float*)alloc(256 * 4);
  float*     psum = (float*)alloc(256 * 4);
  float*     stats= (float*)alloc(2 * 4);
  float*     gpool= (float*)alloc((size_t)BGR * LDIM * 4);
  float*     zsum = (float*)alloc((size_t)BGR * LDIM * 4);
  float*     cnt  = (float*)alloc((size_t)BGR * 4);

  float* out   = (float*)d_out;
  float* pred  = out;
  float* mu    = out + BGR * CDIM;
  float* lv    = mu + (size_t)N_NODES * LDIM;
  float* noise = lv + (size_t)N_NODES * LDIM;

  const int TB = 256;
  auto cdiv = [](long long a, long long b) { return (int)((a + b - 1) / b); };

  // degrees, dinv, per-edge norm
  k_zero<<<cdiv(N_NODES, TB), TB, 0, stream>>>(deg, N_NODES);
  k_deg<<<cdiv(N_EDGES, TB), TB, 0, stream>>>(dst, ea, deg);
  k_dinv<<<cdiv(N_NODES, TB), TB, 0, stream>>>(deg);
  k_norm<<<cdiv(N_EDGES, TB), TB, 0, stream>>>(src, dst, ea, deg, norm);

  // pack weights to f16 WMMA layout
  k_pack_w<<<cdiv(HDIM * HDIM, TB), TB, 0, stream>>>(W1, W1p, HDIM);
  k_pack_w<<<cdiv(HDIM * HDIM, TB), TB, 0, stream>>>(W2, W2p, HDIM);
  k_pack_w<<<cdiv(HDIM * LDIM, TB), TB, 0, stream>>>(Wmu, Wmup, LDIM);
  k_pack_w<<<cdiv(HDIM * LDIM, TB), TB, 0, stream>>>(Wlv, Wlvp, LDIM);

  // encoder input linear
  k_lin_in<<<cdiv((long long)N_NODES * HDIM, TB), TB, 0, stream>>>(x, W_in, b_in, hA);

  const int jobs128 = (N_NODES / 16) * (HDIM / 64);   // 16x64 strip per wave
  const int jobs64  = (N_NODES / 16) * (LDIM / 64);

  // GCN layer 1
  k_gemm_wmma<<<cdiv(jobs128, 8), 256, 0, stream>>>(hA, W1p, b1, hB, HDIM);
  k_zero<<<cdiv((long long)N_NODES * HDIM, TB), TB, 0, stream>>>(hC, N_NODES * HDIM);
  k_scatter<<<cdiv((long long)N_EDGES * (HDIM / 4), TB), TB, 0, stream>>>(hB, src, dst, norm, hC, HDIM);
  k_bn_relu<<<cdiv((long long)N_NODES * HDIM, TB), TB, 0, stream>>>(hB, hC, deg, g1, be1, rm1, rv1, hA);

  // GCN layer 2
  k_gemm_wmma<<<cdiv(jobs128, 8), 256, 0, stream>>>(hA, W2p, b2, hB, HDIM);
  k_zero<<<cdiv((long long)N_NODES * HDIM, TB), TB, 0, stream>>>(hC, N_NODES * HDIM);
  k_scatter<<<cdiv((long long)N_EDGES * (HDIM / 4), TB), TB, 0, stream>>>(hB, src, dst, norm, hC, HDIM);
  k_bn_relu<<<cdiv((long long)N_NODES * HDIM, TB), TB, 0, stream>>>(hB, hC, deg, g2, be2, rm2, rv2, hA);

  // mu head
  k_gemm_wmma<<<cdiv(jobs64, 8), 256, 0, stream>>>(hA, Wmup, bmu, hB, LDIM);
  k_zero<<<cdiv((long long)N_NODES * LDIM, TB), TB, 0, stream>>>(hC, N_NODES * LDIM);
  k_scatter<<<cdiv((long long)N_EDGES * (LDIM / 4), TB), TB, 0, stream>>>(hB, src, dst, norm, hC, LDIM);
  k_selfloop<<<cdiv((long long)N_NODES * LDIM, TB), TB, 0, stream>>>(hB, hC, deg, mu);

  // logvar head
  k_gemm_wmma<<<cdiv(jobs64, 8), 256, 0, stream>>>(hA, Wlvp, blv, hB, LDIM);
  k_zero<<<cdiv((long long)N_NODES * LDIM, TB), TB, 0, stream>>>(hC, N_NODES * LDIM);
  k_scatter<<<cdiv((long long)N_EDGES * (LDIM / 4), TB), TB, 0, stream>>>(hB, src, dst, norm, hC, LDIM);
  k_selfloop<<<cdiv((long long)N_NODES * LDIM, TB), TB, 0, stream>>>(hB, hC, deg, lv);

  // attention pooling with global softmax over all nodes
  k_att<<<N_NODES, HDIM, 0, stream>>>(mu, Wa1, ba1, Wa2, ba2, score);
  k_rmax<<<256, 256, 0, stream>>>(score, N_NODES, pmax);
  k_rmax<<<1, 256, 0, stream>>>(pmax, 256, stats);            // stats[0] = max
  k_rsumexp<<<256, 256, 0, stream>>>(score, N_NODES, stats, psum);
  k_rsum<<<1, 256, 0, stream>>>(psum, 256, stats + 1);        // stats[1] = sum(exp)
  k_zero<<<cdiv(BGR * LDIM, TB), TB, 0, stream>>>(gpool, BGR * LDIM);
  k_zero<<<cdiv(BGR * LDIM, TB), TB, 0, stream>>>(zsum, BGR * LDIM);
  k_zero<<<1, TB, 0, stream>>>(cnt, BGR);
  k_pool<<<cdiv((long long)N_NODES * LDIM, TB), TB, 0, stream>>>(mu, score, batch, stats, gpool, zsum, cnt);

  // per-graph heads
  k_head<<<BGR, HDIM, 0, stream>>>(gpool, zsum, cnt, Wc1, bc1, Wc2, bc2, Wc3, bc3,
                                   Wn1, bn1, Wn2, bn2, pred, noise);
}
